// Heat2D_65180423685023
// MI455X (gfx1250) — compile-verified, matrix operand order checked by reading, not run on hardware
//
#include <hip/hip_runtime.h>
#include <hip/hip_bf16.h>
#include <math.h>

// ---------------- CDNA5 WMMA types ----------------
typedef __attribute__((ext_vector_type(16))) __bf16 v16bf;
typedef __attribute__((ext_vector_type(8)))  float  v8f;
typedef __attribute__((ext_vector_type(2)))  float  v2f;

#if defined(__has_builtin)
#if __has_builtin(__builtin_amdgcn_wmma_f32_16x16x4_f32)
#define HAS_F32_WMMA 1
#endif
#endif

#define NB   64      // batch
#define NC   96      // channels
#define NH   56
#define NW   56
#define NHW  3136    // 56*56
#define PI_D 3.14159265358979323846

// =====================================================================
// 1) Constant matrices: Wp (DCT cos map), Wpt, Cp (cos 2pi nh/56), Sp
//    All padded to 64x64 with zeros.
// =====================================================================
__global__ __launch_bounds__(256) void k_mats(float* __restrict__ Wp, float* __restrict__ Wpt,
                                              float* __restrict__ Cp, float* __restrict__ Sp) {
  for (int e = threadIdx.x; e < 4096; e += 256) {
    const int n = e >> 6, h = e & 63;
    float wv = 0.f, cv = 0.f, sv = 0.f;
    if (n < 56 && h < 56) {
      double v = cos((double)n * (((double)h + 0.5) / 56.0) * PI_D) * sqrt(2.0 / 56.0);
      if (n == 0) v *= 0.70710678118654752440;
      wv = (float)v;
      const double ang = 2.0 * PI_D * (double)((n * h) % 56) / 56.0;
      cv = (float)cos(ang);
      sv = (float)sin(ang);
    }
    Wp[e] = wv;
    Wpt[h * 64 + n] = wv;
    Cp[e] = cv;   // symmetric: Cp == Cp^T
    Sp[e] = sv;   // symmetric: Sp == Sp^T
  }
}

// =====================================================================
// 2) wexp_p[c][n][m] = exp(-(an^2+bm^2) * relu(freq_embed[n,m,:]@tok_w[c,:]+tok_b[c]))
// =====================================================================
__global__ __launch_bounds__(256) void k_wexp(const float* __restrict__ fe,
                                              const float* __restrict__ tok_w,
                                              const float* __restrict__ tok_b,
                                              float* __restrict__ wexp_p) {
  const int idx = blockIdx.x * 256 + threadIdx.x;
  if (idx >= NC * NHW) return;
  const int c = idx / NHW, hw = idx - c * NHW;
  const int n = hw / NW, m = hw - n * NW;
  const float* fr = fe + (size_t)hw * NC;
  const float* tr = tok_w + c * NC;
  float k = tok_b[c];
  for (int j = 0; j < NC; ++j) k += fr[j] * tr[j];
  k = fmaxf(k, 0.f);
  const float an = (float)n * (float)(PI_D / 56.0);
  const float bm = (float)m * (float)(PI_D / 56.0);
  wexp_p[(size_t)c * NHW + hw] = __expf(-(an * an + bm * bm) * k);
}

// =====================================================================
// 3) B-operand fragment prep: swizzle lin_w (192x96) and out_w (96x96)
//    into per-wave WMMA bf16 B-fragments (32 lanes x 16 bf16 each).
//    Lane l: n = l&15 ; k = k0 + 8*(l>>4) + (i<8 ? i : i+8)
// =====================================================================
__global__ __launch_bounds__(256) void k_prep(const float* __restrict__ lin_w,
                                              const float* __restrict__ out_w,
                                              __bf16* __restrict__ linwf,
                                              __bf16* __restrict__ outwf) {
  const int E1 = 3 * 12 * 32 * 16;  // lin_w: KT=3, NT=12
  const int E2 = 3 * 6 * 32 * 16;   // out_w: KT=3, NT=6
  const int idx = blockIdx.x * 256 + threadIdx.x;
  if (idx < E1) {
    const int i = idx & 15, l = (idx >> 4) & 31, j = idx >> 9;
    const int kt = j / 12, nt = j - kt * 12;
    const int kk = kt * 32 + 8 * (l >> 4) + (i < 8 ? i : i + 8);
    linwf[idx] = (__bf16)lin_w[(nt * 16 + (l & 15)) * NC + kk];
  } else if (idx < E1 + E2) {
    const int e = idx - E1;
    const int i = e & 15, l = (e >> 4) & 31, j = e >> 9;
    const int kt = j / 6, nt = j - kt * 6;
    const int kk = kt * 32 + 8 * (l >> 4) + (i < 8 ? i : i + 8);
    outwf[e] = (__bf16)out_w[(nt * 16 + (l & 15)) * NC + kk];
  }
}

// =====================================================================
// 4) Depthwise 3x3 SAME conv -> bf16 token-major activations xc_t[(b*3136+hw)*96 + c]
// =====================================================================
__global__ __launch_bounds__(256) void k_dwconv(const float* __restrict__ x,
                                                const float* __restrict__ dw_w,
                                                const float* __restrict__ dw_b,
                                                __bf16* __restrict__ xc_t) {
  __shared__ __align__(16) __bf16 tt[64 * 100];
  const int blk = blockIdx.x;
  const int b = blk / 49;
  const int hw0 = (blk - b * 49) * 64;
  for (int e = threadIdx.x; e < NC * 64; e += 256) {
    const int c = e >> 6, i = e & 63;
    const int hw = hw0 + i;
    const int h = hw / NW, w = hw - h * NW;
    const float* xp = x + (size_t)(b * NC + c) * NHW;
    const float* kp = dw_w + c * 9;
    float acc = dw_b[c];
#pragma unroll
    for (int dy = -1; dy <= 1; ++dy) {
      const int hh = h + dy;
      if (hh < 0 || hh >= NH) continue;
#pragma unroll
      for (int dx = -1; dx <= 1; ++dx) {
        const int ww = w + dx;
        if (ww < 0 || ww >= NW) continue;
        acc += xp[hh * NW + ww] * kp[(dy + 1) * 3 + (dx + 1)];
      }
    }
    tt[i * 100 + c] = (__bf16)acc;
  }
  __syncthreads();
  __bf16* dst = xc_t + ((size_t)b * NHW + hw0) * NC;
  for (int e = threadIdx.x; e < NC * 64; e += 256) {
    const int i = e / NC, c = e - i * NC;
    dst[e] = tt[i * 100 + c];  // e == i*96 + c : contiguous global write
  }
}

// =====================================================================
// 5) GEMM1 (bf16 WMMA): y[tok, d] = xc_t[tok,:] @ lin_w[d,:] + lin_b[d]
//    d<96 -> x1 (plane-major fp32), d>=96 -> silu(z) token-major fp32.
//    Block: 32 tokens (same b), 8 waves. Wave owns mt = wave&1, walks 3
//    N-tiles (base, base+4, base+8) reusing its preloaded A fragments.
// =====================================================================
__global__ __launch_bounds__(256) void k_gemm1(const __bf16* __restrict__ xc_t,
                                               const __bf16* __restrict__ linwf,
                                               const float* __restrict__ lin_b,
                                               float* __restrict__ x1,
                                               float* __restrict__ z_act) {
  __shared__ __align__(16) __bf16 ga[32 * 96];
  __shared__ float yt[32 * 193];
  const int blk = blockIdx.x;
  const int b = blk / 98;
  const int hw0 = (blk - b * 98) * 32;
  const size_t tok0 = (size_t)b * NHW + hw0;

  // Warm L2 for the reused B-fragment table (gfx1250 global_prefetch_b8).
  __builtin_prefetch(linwf + (size_t)threadIdx.x * 72, 0, 1);

  {  // stage A tile (contiguous 6144B)
    const uint32_t* src = (const uint32_t*)(xc_t + tok0 * NC);
    uint32_t* dst = (uint32_t*)ga;
    for (int e = threadIdx.x; e < 32 * 96 / 2; e += 256) dst[e] = src[e];
  }
  __syncthreads();

  const int lane = threadIdx.x & 31, wave = threadIdx.x >> 5;
  const int mrow = lane & 15, hi = lane >> 4;
  const int mt = wave & 1;
  const int ntbase = wave >> 1;  // 0..3

  // Preload this wave's A fragments once (reused for 3 N-tiles).
  v16bf afr[3];
#pragma unroll
  for (int kt = 0; kt < 3; ++kt) {
    const __bf16* row = ga + (mt * 16 + mrow) * 96 + kt * 32 + 8 * hi;
#pragma unroll
    for (int i2 = 0; i2 < 8; ++i2) { afr[kt][i2] = row[i2]; afr[kt][i2 + 8] = row[i2 + 16]; }
  }

#pragma unroll
  for (int jj = 0; jj < 3; ++jj) {
    const int nt = ntbase + jj * 4;  // covers 0..11 across waves
    v8f acc;
    for (int r = 0; r < 8; ++r) acc[r] = 0.f;
#pragma unroll
    for (int kt = 0; kt < 3; ++kt) {
      const v16bf bv = *(const v16bf*)(linwf + (size_t)((kt * 12 + nt) * 32 + lane) * 16);
      acc = __builtin_amdgcn_wmma_f32_16x16x32_bf16(false, afr[kt], false, bv, (short)0, acc, false, false);
    }
    const int col = nt * 16 + mrow;
    const float bias = lin_b[col];
#pragma unroll
    for (int r = 0; r < 8; ++r) yt[(mt * 16 + r + 8 * hi) * 193 + col] = acc[r] + bias;
  }
  __syncthreads();

  // x1 half -> plane-major (b,c,hw)
  for (int e = threadIdx.x; e < NC * 32; e += 256) {
    const int c = e >> 5, i = e & 31;
    x1[(size_t)(b * NC + c) * NHW + hw0 + i] = yt[i * 193 + c];
  }
  // z half -> silu, token-major (b,hw,c)
  for (int e = threadIdx.x; e < NC * 32; e += 256) {
    const int i = e / NC, c = e - i * NC;
    const float z = yt[i * 193 + 96 + c];
    z_act[(tok0 + i) * NC + c] = z / (1.f + __expf(-z));
  }
}

// =====================================================================
// 6) Deterministic spatial-mean pooling: pool[b*96+c] = sum(x1 plane)
// =====================================================================
__global__ __launch_bounds__(256) void k_pool(const float* __restrict__ x1, float* __restrict__ pool) {
  __shared__ float red[256];
  const float* plane = x1 + (size_t)blockIdx.x * NHW;
  float s = 0.f;
  for (int i = threadIdx.x; i < NHW; i += 256) s += plane[i];
  red[threadIdx.x] = s;
  __syncthreads();
  for (int st = 128; st > 0; st >>= 1) {
    if (threadIdx.x < st) red[threadIdx.x] += red[threadIdx.x + st];
    __syncthreads();
  }
  if (threadIdx.x == 0) pool[blockIdx.x] = red[0];
}

// =====================================================================
// 7) MoE head: mean -> MLP(96->256->32->3) with lrelu+BN, +gumbel, argmax
// =====================================================================
__global__ __launch_bounds__(256) void k_moe(const float* __restrict__ pool,
    const float* __restrict__ p1_w, const float* __restrict__ p1_b,
    const float* __restrict__ g1, const float* __restrict__ bb1,
    const float* __restrict__ m1, const float* __restrict__ v1,
    const float* __restrict__ p2_w, const float* __restrict__ p2_b,
    const float* __restrict__ g2, const float* __restrict__ bb2,
    const float* __restrict__ m2, const float* __restrict__ v2,
    const float* __restrict__ p3_w, const float* __restrict__ p3_b,
    const float* __restrict__ gumbel, int* __restrict__ branch) {
  __shared__ float meanv[96], h1[256], h2[32], lg[3];
  const int b = blockIdx.x, t = threadIdx.x;
  if (t < 96) meanv[t] = pool[b * 96 + t] * (1.f / (float)NHW);
  __syncthreads();
  {
    float a = p1_b[t];
    const float* wr = p1_w + t * 96;
    for (int c = 0; c < 96; ++c) a += meanv[c] * wr[c];
    a = (a >= 0.f) ? a : 0.1f * a;
    h1[t] = (a - m1[t]) * rsqrtf(v1[t] + 1e-5f) * g1[t] + bb1[t];
  }
  __syncthreads();
  if (t < 32) {
    float a = p2_b[t];
    const float* wr = p2_w + t * 256;
    for (int j = 0; j < 256; ++j) a += h1[j] * wr[j];
    a = (a >= 0.f) ? a : 0.1f * a;
    h2[t] = (a - m2[t]) * rsqrtf(v2[t] + 1e-5f) * g2[t] + bb2[t];
  }
  __syncthreads();
  if (t < 3) {
    float a = p3_b[t];
    const float* wr = p3_w + t * 32;
    for (int j = 0; j < 32; ++j) a += h2[j] * wr[j];
    lg[t] = a + gumbel[b * 3 + t];
  }
  __syncthreads();
  if (t == 0) {
    int best = 0;
    float bv = lg[0];
    if (lg[1] > bv) { bv = lg[1]; best = 1; }
    if (lg[2] > bv) { best = 2; }
    branch[b] = best;
  }
}

// =====================================================================
// 8) fp32 WMMA 64x64x64 matmul helper (8 waves, 16 tiles, K=64).
//    D = (doacc ? D : 0) + alpha * A*B.  A/B/D row-major stride 64.
// =====================================================================
__device__ __forceinline__ void mm64(const float* __restrict__ A, const float* __restrict__ Bm,
                                     float* __restrict__ D, float alpha, bool doacc) {
  const int lane = threadIdx.x & 31;
  const int wave = threadIdx.x >> 5;
#if defined(HAS_F32_WMMA)
  const int mrow = lane & 15;
  const int hi = lane >> 4;
  const int kb = hi * 2;
  for (int t = wave; t < 16; t += 8) {
    const int tm = (t >> 2) * 16, tn = (t & 3) * 16;
    v8f c;
    for (int r = 0; r < 8; ++r) c[r] = 0.f;
#pragma unroll 4
    for (int k = 0; k < 64; k += 4) {
      v2f a, b;
      const float* ar = A + (tm + mrow) * 64 + k + kb;
      a[0] = ar[0];
      a[1] = ar[1];
      const float* br = Bm + (k + kb) * 64 + tn + mrow;
      b[0] = br[0];
      b[1] = br[64];
      c = __builtin_amdgcn_wmma_f32_16x16x4_f32(false, a, false, b, (short)0, c, false, false);
    }
    for (int r = 0; r < 8; ++r) {
      float* dp = D + (tm + r + 8 * hi) * 64 + tn + mrow;
      float v = alpha * c[r];
      if (doacc) v += *dp;
      *dp = v;
    }
  }
#else
  (void)wave; (void)lane;
  for (int e = threadIdx.x; e < 4096; e += 256) {
    const int i = e >> 6, j = e & 63;
    float s = 0.f;
    for (int k = 0; k < 64; ++k) s += A[i * 64 + k] * Bm[k * 64 + j];
    float v = alpha * s;
    if (doacc) v += D[e];
    D[e] = v;
  }
#endif
}

// =====================================================================
// 9) Spectral filter, one (b,c) plane per block, in-place on x1.
//    branch 0: DCT   1: FFT   2: Haar
//    Dynamic LDS: 4 x 64x64 fp32 buffers = 65536 B.
// =====================================================================
__global__ __launch_bounds__(256) void k_spectral(float* __restrict__ x1,
                                                  const float* __restrict__ wexp_p,
                                                  const int* __restrict__ branch,
                                                  const float* __restrict__ Wp,
                                                  const float* __restrict__ Wpt,
                                                  const float* __restrict__ Cp,
                                                  const float* __restrict__ Sp) {
  extern __shared__ float sm[];
  float* S  = sm;
  float* B1 = sm + 4096;
  float* B2 = sm + 8192;
  float* B3 = sm + 12288;
  const int blk = blockIdx.x;
  const int b = blk / NC, c = blk - b * NC;
  float* plane = x1 + (size_t)blk * NHW;
  const float* wg = wexp_p + (size_t)c * NHW;
  const int br = branch[b];

  // Warm L2 for the filter plane while we zero/stage LDS.
  __builtin_prefetch(wg + (size_t)threadIdx.x * 12, 0, 1);

  for (int e = threadIdx.x; e < 4096; e += 256) S[e] = 0.f;
  __syncthreads();
  for (int e = threadIdx.x; e < NHW; e += 256) {
    const int h = e / NW, w = e - h * NW;
    S[h * 64 + w] = plane[e];
  }
  __syncthreads();

  if (br == 0) {  // ---- DCT branch ----
    mm64(Wp, S, B1, 1.f, false);  __syncthreads();
    mm64(B1, Wpt, B2, 1.f, false); __syncthreads();
    for (int e = threadIdx.x; e < NHW; e += 256) {
      const int h = e / NW, w = e - h * NW;
      B2[h * 64 + w] *= wg[e];
    }
    __syncthreads();
    mm64(Wpt, B2, B1, 1.f, false); __syncthreads();
    mm64(B1, Wp, B3, 1.f, false);  __syncthreads();
    for (int e = threadIdx.x; e < NHW; e += 256) {
      const int h = e / NW, w = e - h * NW;
      plane[e] = B3[h * 64 + w];
    }
  } else if (br == 1) {  // ---- FFT branch (C,S symmetric) ----
    mm64(Cp, S, B1, 1.f, false);   __syncthreads();  // U1 = C s
    mm64(Sp, S, B2, 1.f, false);   __syncthreads();  // U2 = S s
    mm64(B1, Cp, S, 1.f, false);   __syncthreads();  // P  = U1 C
    mm64(B2, Sp, S, -1.f, true);   __syncthreads();  //    - U2 S
    mm64(B1, Sp, B3, 1.f, false);  __syncthreads();  // Q  = U1 S
    mm64(B2, Cp, B3, 1.f, true);   __syncthreads();  //    + U2 C
    const float inv = 1.f / (float)NHW;
    for (int e = threadIdx.x; e < NHW; e += 256) {
      const int h = e / NW, w = e - h * NW;
      const float wf = wg[e] * inv;
      S[h * 64 + w]  *= wf;   // Yr
      B3[h * 64 + w] *= wf;   // Yi
    }
    __syncthreads();
    mm64(S, Cp, B1, 1.f, false);   __syncthreads();  // R1 = Yr C
    mm64(B3, Sp, B1, 1.f, true);   __syncthreads();  //    + Yi S
    mm64(B3, Cp, B2, 1.f, false);  __syncthreads();  // R2 = Yi C
    mm64(S, Sp, B2, -1.f, true);   __syncthreads();  //    - Yr S
    mm64(Cp, B1, S, 1.f, false);   __syncthreads();  // out = C R1
    mm64(Sp, B2, S, 1.f, true);    __syncthreads();  //     + S R2
    for (int e = threadIdx.x; e < NHW; e += 256) {
      const int h = e / NW, w = e - h * NW;
      plane[e] = S[h * 64 + w];
    }
  } else {  // ---- Haar branch ----
    for (int e = threadIdx.x; e < NHW; e += 256) {  // haar along H
      const int i = e / NW, w = e - (e / NW) * NW;
      float v;
      if (i < 28) v = 0.5f * (S[(2 * i) * 64 + w] + S[(2 * i + 1) * 64 + w]);
      else { const int k = i - 28; v = 0.5f * (S[(2 * k) * 64 + w] - S[(2 * k + 1) * 64 + w]); }
      B1[i * 64 + w] = v;
    }
    __syncthreads();
    for (int e = threadIdx.x; e < NHW; e += 256) {  // haar along W, * wexp
      const int h = e / NW, j = e - (e / NW) * NW;
      float v;
      if (j < 28) v = 0.5f * (B1[h * 64 + 2 * j] + B1[h * 64 + 2 * j + 1]);
      else { const int k = j - 28; v = 0.5f * (B1[h * 64 + 2 * k] - B1[h * 64 + 2 * k + 1]); }
      B2[h * 64 + j] = v * wg[e];
    }
    __syncthreads();
    for (int e = threadIdx.x; e < NHW; e += 256) {  // inverse haar along H
      const int h = e / NW, w = e - (e / NW) * NW;
      const int k = h >> 1;
      const float av = B2[k * 64 + w], dv = B2[(28 + k) * 64 + w];
      B1[h * 64 + w] = (h & 1) ? (av - dv) : (av + dv);
    }
    __syncthreads();
    for (int e = threadIdx.x; e < NHW; e += 256) {  // inverse haar along W -> out
      const int h = e / NW, w = e - (e / NW) * NW;
      const int k = w >> 1;
      const float av = B1[h * 64 + k], dv = B1[h * 64 + 28 + k];
      plane[e] = (w & 1) ? (av - dv) : (av + dv);
    }
  }
}

// =====================================================================
// 10) Final: LayerNorm(C) + silu(z) gate + GEMM2 (bf16 WMMA, 96x96) + NCHW store
// =====================================================================
__global__ __launch_bounds__(256) void k_final(const float* __restrict__ xs,
                                               const float* __restrict__ z_act,
                                               const float* __restrict__ ln_g,
                                               const float* __restrict__ ln_b,
                                               const __bf16* __restrict__ outwf,
                                               const float* __restrict__ out_b,
                                               float* __restrict__ out) {
  __shared__ float xt[32 * 97];
  __shared__ __align__(16) __bf16 g[32 * 96];
  __shared__ float ot[32 * 97];
  __shared__ float mu[32], rs[32];
  const int blk = blockIdx.x;
  const int b = blk / 98;
  const int hw0 = (blk - b * 98) * 32;

  __builtin_prefetch(outwf + (size_t)threadIdx.x * 36, 0, 1);

  for (int e = threadIdx.x; e < NC * 32; e += 256) {
    const int c = e >> 5, i = e & 31;
    xt[i * 97 + c] = xs[(size_t)(b * NC + c) * NHW + hw0 + i];
  }
  __syncthreads();
  if (threadIdx.x < 32) {
    const int i = threadIdx.x;
    float s = 0.f;
    for (int c = 0; c < NC; ++c) s += xt[i * 97 + c];
    const float m = s * (1.f / (float)NC);
    float v = 0.f;
    for (int c = 0; c < NC; ++c) { const float d = xt[i * 97 + c] - m; v += d * d; }
    mu[i] = m;
    rs[i] = rsqrtf(v * (1.f / (float)NC) + 1e-5f);
  }
  __syncthreads();
  for (int e = threadIdx.x; e < 32 * NC; e += 256) {
    const int i = e / NC, c = e - i * NC;
    float val = (xt[i * 97 + c] - mu[i]) * rs[i] * ln_g[c] + ln_b[c];
    val *= z_act[((size_t)b * NHW + hw0 + i) * NC + c];
    g[i * NC + c] = (__bf16)val;
  }
  __syncthreads();

  const int lane = threadIdx.x & 31, wave = threadIdx.x >> 5;
  const int mrow = lane & 15, hi = lane >> 4;
  const int mt = wave & 1;
  const int ntbase = wave >> 1;  // 0..3

  // Preload A fragments for this wave's M-tile.
  v16bf afr[3];
#pragma unroll
  for (int kt = 0; kt < 3; ++kt) {
    const __bf16* row = g + (mt * 16 + mrow) * NC + kt * 32 + 8 * hi;
#pragma unroll
    for (int i2 = 0; i2 < 8; ++i2) { afr[kt][i2] = row[i2]; afr[kt][i2 + 8] = row[i2 + 16]; }
  }

  for (int nt = ntbase; nt < 6; nt += 4) {  // waves 0..3 do 2 tiles, 4..7 do 1
    v8f acc;
    for (int r = 0; r < 8; ++r) acc[r] = 0.f;
#pragma unroll
    for (int kt = 0; kt < 3; ++kt) {
      const v16bf bv = *(const v16bf*)(outwf + (size_t)((kt * 6 + nt) * 32 + lane) * 16);
      acc = __builtin_amdgcn_wmma_f32_16x16x32_bf16(false, afr[kt], false, bv, (short)0, acc, false, false);
    }
    const int col = nt * 16 + mrow;
    const float bias = out_b[col];
#pragma unroll
    for (int r = 0; r < 8; ++r) ot[(mt * 16 + r + 8 * hi) * 97 + col] = acc[r] + bias;
  }
  __syncthreads();
  for (int e = threadIdx.x; e < NC * 32; e += 256) {
    const int n = e >> 5, i = e & 31;
    out[(size_t)(b * NC + n) * NHW + hw0 + i] = ot[i * 97 + n];
  }
}

// =====================================================================
// Host launcher
// =====================================================================
extern "C" void kernel_launch(void* const* d_in, const int* in_sizes, int n_in,
                              void* d_out, int out_size, void* d_ws, size_t ws_size,
                              hipStream_t stream) {
  (void)in_sizes; (void)n_in; (void)out_size; (void)ws_size;
  const float* x     = (const float*)d_in[0];
  const float* freq  = (const float*)d_in[1];
  const float* gum   = (const float*)d_in[2];
  const float* dw_w  = (const float*)d_in[3];
  const float* dw_b  = (const float*)d_in[4];
  const float* lin_w = (const float*)d_in[5];
  const float* lin_b = (const float*)d_in[6];
  const float* p1_w  = (const float*)d_in[7];
  const float* p1_b  = (const float*)d_in[8];
  const float* bn1_g = (const float*)d_in[9];
  const float* bn1_b = (const float*)d_in[10];
  const float* bn1_m = (const float*)d_in[11];
  const float* bn1_v = (const float*)d_in[12];
  const float* p2_w  = (const float*)d_in[13];
  const float* p2_b  = (const float*)d_in[14];
  const float* bn2_g = (const float*)d_in[15];
  const float* bn2_b = (const float*)d_in[16];
  const float* bn2_m = (const float*)d_in[17];
  const float* bn2_v = (const float*)d_in[18];
  const float* p3_w  = (const float*)d_in[19];
  const float* p3_b  = (const float*)d_in[20];
  const float* tok_w = (const float*)d_in[21];
  const float* tok_b = (const float*)d_in[22];
  const float* ln_g  = (const float*)d_in[23];
  const float* ln_b  = (const float*)d_in[24];
  const float* out_w = (const float*)d_in[25];
  const float* out_b = (const float*)d_in[26];
  float* outp = (float*)d_out;

  // ---- workspace partition ----
  char* p = (char*)d_ws;
  auto alloc = [&](size_t bytes) -> char* {
    char* r = p;
    p += (bytes + 255) & ~(size_t)255;
    return r;
  };
  __bf16* xc_t   = (__bf16*)alloc((size_t)NB * NHW * NC * 2);
  float*  x1     = (float*) alloc((size_t)NB * NC * NHW * 4);
  float*  z_act  = (float*) alloc((size_t)NB * NHW * NC * 4);
  float*  pool   = (float*) alloc((size_t)NB * NC * 4);
  int*    branch = (int*)   alloc((size_t)NB * 4);
  float*  wexp_p = (float*) alloc((size_t)NC * NHW * 4);
  float*  Wp     = (float*) alloc(4096 * 4);
  float*  Wpt    = (float*) alloc(4096 * 4);
  float*  Cp     = (float*) alloc(4096 * 4);
  float*  Sp     = (float*) alloc(4096 * 4);
  __bf16* linwf  = (__bf16*)alloc((size_t)3 * 12 * 32 * 16 * 2);
  __bf16* outwf  = (__bf16*)alloc((size_t)3 * 6 * 32 * 16 * 2);

  k_mats<<<1, 256, 0, stream>>>(Wp, Wpt, Cp, Sp);
  k_wexp<<<(NC * NHW + 255) / 256, 256, 0, stream>>>(freq, tok_w, tok_b, wexp_p);
  k_prep<<<(3 * 12 * 32 * 16 + 3 * 6 * 32 * 16) / 256, 256, 0, stream>>>(lin_w, out_w, linwf, outwf);
  k_dwconv<<<NB * 49, 256, 0, stream>>>(x, dw_w, dw_b, xc_t);
  k_gemm1<<<NB * 98, 256, 0, stream>>>(xc_t, linwf, lin_b, x1, z_act);
  k_pool<<<NB * NC, 256, 0, stream>>>(x1, pool);
  k_moe<<<NB, 256, 0, stream>>>(pool, p1_w, p1_b, bn1_g, bn1_b, bn1_m, bn1_v,
                                p2_w, p2_b, bn2_g, bn2_b, bn2_m, bn2_v,
                                p3_w, p3_b, gum, branch);
  k_spectral<<<NB * NC, 256, 65536, stream>>>(x1, wexp_p, branch, Wp, Wpt, Cp, Sp);
  k_final<<<NB * 98, 256, 0, stream>>>(x1, z_act, ln_g, ln_b, outwf, out_b, outp);
}
